// MultiHeadSelfAttention_63934883168488
// MI455X (gfx1250) — matrix operand order repaired and checked by reading, not compiled, not run on hardware
//
#include <hip/hip_runtime.h>

typedef unsigned short u16t;
typedef unsigned int   u32t;
typedef __attribute__((ext_vector_type(16))) __bf16 v16bf;
typedef __attribute__((ext_vector_type(8)))  float  v8f;

static constexpr int Bc = 4, Sc = 2048, Dc = 2048, Hc = 16, DHc = 128;
static constexpr float SCALE = 0.08838834764831845f; // 1/sqrt(128)

union Frag { v16bf v; uint4 q[2]; u16t u[16]; };
union Acc  { v8f v; float f[8]; };

__device__ __forceinline__ void sched_fence() {
#if __has_builtin(__builtin_amdgcn_sched_barrier)
  __builtin_amdgcn_sched_barrier(0);
#else
  asm volatile("" ::: "memory");
#endif
}

__device__ __forceinline__ u16t f2bf(float f) {
  u32t u = __float_as_uint(f);
  u32t r = u + 0x7FFFu + ((u >> 16) & 1u);   // round-to-nearest-even
  return (u16t)(r >> 16);
}
__device__ __forceinline__ float bf2f(u16t h) {
  return __uint_as_float(((u32t)h) << 16);
}
__device__ __forceinline__ u32t pack2(float a, float b) {
  return (u32t)f2bf(a) | ((u32t)f2bf(b) << 16);
}
__device__ __forceinline__ v8f wmma_bf16(v16bf a, v16bf b, v8f c) {
  return __builtin_amdgcn_wmma_f32_16x16x32_bf16(false, a, false, b, (short)0, c, false, false);
}

// CDNA5 async global->LDS copy (16B per lane), tracked by ASYNCcnt.
__device__ __forceinline__ void async_ld_b128(const u16t* gaddr, u32t lds_off) {
  asm volatile("global_load_async_to_lds_b128 %0, %1, off"
               :: "v"(lds_off), "v"(gaddr) : "memory");
}
__device__ __forceinline__ void wait_async_2() {
#if __has_builtin(__builtin_amdgcn_s_wait_asynccnt)
  __builtin_amdgcn_s_wait_asynccnt(2);
#else
  asm volatile("s_wait_asynccnt 0x2" ::: "memory");
#endif
}
__device__ __forceinline__ void wait_async_0() {
#if __has_builtin(__builtin_amdgcn_s_wait_asynccnt)
  __builtin_amdgcn_s_wait_asynccnt(0);
#else
  asm volatile("s_wait_asynccnt 0x0" ::: "memory");
#endif
}

// ---------------- fp32 -> bf16 convert ----------------
__global__ void k_cvt_bf16(const float* __restrict__ src, u16t* __restrict__ dst, int n) {
  int i = blockIdx.x * blockDim.x + threadIdx.x;
  if (i < n) dst[i] = f2bf(src[i]);
}

// ---------------- GEMM: Y = X * W^T (both bf16 row-major, K = 2048) ----------------
// MODE 0: bf16 out (b,h,s,d)   MODE 1: bf16 out (b,h,d,s)   MODE 2: fp32 out (row,col)
static constexpr int LDS_ROW = 40;                 // u16 elems (32 data + 8 pad) = 80B
static constexpr u32t LDS_BUF = 128 * LDS_ROW * 2; // bytes per buffer = 10240

template<int MODE>
__global__ __launch_bounds__(256, 1) void k_gemm_nt(const u16t* __restrict__ X,
                                                    const u16t* __restrict__ W,
                                                    void* __restrict__ outp) {
  __shared__ u16t Bl[2][128 * LDS_ROW];

  const int tid = threadIdx.x;
  const int lane = tid & 31;
  const int wave = tid >> 5;
  const int lo = lane & 15;
  const int hi = lane >> 4;
  const int Mbase = blockIdx.y * 128 + wave * 16;
  const int Nstrip = blockIdx.x * 128;

  const u16t* arow = X + (size_t)(Mbase + lo) * Dc;

  // async-copy mapping: 512 x b128 per 128x32 tile; 2 per thread
  const int i0 = tid * 2, i1 = tid * 2 + 1;
  const int r0 = i0 >> 2, p0 = i0 & 3;
  const int r1 = i1 >> 2, p1 = i1 & 3;
  const u16t* g0 = W + (size_t)(Nstrip + r0) * Dc + p0 * 8;
  const u16t* g1 = W + (size_t)(Nstrip + r1) * Dc + p1 * 8;
  const u32t lbase = (u32t)(uintptr_t)(&Bl[0][0]);
  const u32t l0 = lbase + (u32t)(r0 * LDS_ROW * 2 + p0 * 16);
  const u32t l1 = lbase + (u32t)(r1 * LDS_ROW * 2 + p1 * 16);

  async_ld_b128(g0, l0);
  async_ld_b128(g1, l1);

  Acc acc[8];
#pragma unroll
  for (int i = 0; i < 8; ++i)
#pragma unroll
    for (int r = 0; r < 8; ++r) acc[i].f[r] = 0.0f;

  Frag a;
  a.q[0] = *(const uint4*)(arow + 8 * hi);
  a.q[1] = *(const uint4*)(arow + 16 + 8 * hi);

  constexpr int STEPS = Dc / 32;   // 64
  for (int i = 0; i < STEPS; ++i) {
    const int kb = i * 32;
    const int cur = i & 1;
    const int nxt = cur ^ 1;

    __syncthreads();               // all waves done reading buf[nxt] (iter i-1)

    Frag an = a;
    if (i + 1 < STEPS) {
      an.q[0] = *(const uint4*)(arow + kb + 32 + 8 * hi);
      an.q[1] = *(const uint4*)(arow + kb + 32 + 16 + 8 * hi);
      async_ld_b128(g0 + kb + 32, l0 + (u32t)nxt * LDS_BUF);
      async_ld_b128(g1 + kb + 32, l1 + (u32t)nxt * LDS_BUF);
      wait_async_2();              // my buf[cur] transfers done, prefetch in flight
    } else {
      wait_async_0();
    }
    __syncthreads();               // buf[cur] visible to all waves

    // stage all 8 B fragments from LDS, then WMMAs back-to-back
    Frag b[8];
#pragma unroll
    for (int nt = 0; nt < 8; ++nt) {
      b[nt].q[0] = *(const uint4*)(&Bl[cur][(nt * 16 + lo) * LDS_ROW + 8 * hi]);
      b[nt].q[1] = *(const uint4*)(&Bl[cur][(nt * 16 + lo) * LDS_ROW + 16 + 8 * hi]);
    }
    sched_fence();                 // keep all ds_loads issued before the WMMA burst
#pragma unroll
    for (int nt = 0; nt < 8; ++nt)
      acc[nt].v = wmma_bf16(a.v, b[nt].v, acc[nt].v);
    a = an;
  }

  // C/D layout: element (m,n): vgpr r = m%8, lane = n + 16*(m/8)
  if (MODE == 0) {
    u16t* out = (u16t*)outp;
#pragma unroll
    for (int nt = 0; nt < 8; ++nt) {
      int o = Nstrip + nt * 16 + lo;
      int h = o >> 7, d = o & 127;
#pragma unroll
      for (int r = 0; r < 8; ++r) {
        int t = Mbase + 8 * hi + r;
        int bb = t >> 11, s = t & (Sc - 1);
        out[((size_t)(bb * Hc + h) * Sc + s) * DHc + d] = f2bf(acc[nt].f[r]);
      }
    }
  } else if (MODE == 1) {
    u16t* out = (u16t*)outp;
    int t0 = Mbase + 8 * hi;
    int bb = t0 >> 11, s0 = t0 & (Sc - 1);
#pragma unroll
    for (int nt = 0; nt < 8; ++nt) {
      int o = Nstrip + nt * 16 + lo;
      int h = o >> 7, d = o & 127;
      uint4 w;
      w.x = pack2(acc[nt].f[0], acc[nt].f[1]);
      w.y = pack2(acc[nt].f[2], acc[nt].f[3]);
      w.z = pack2(acc[nt].f[4], acc[nt].f[5]);
      w.w = pack2(acc[nt].f[6], acc[nt].f[7]);
      *(uint4*)(out + ((size_t)(bb * Hc + h) * DHc + d) * Sc + s0) = w;
    }
  } else {
    float* out = (float*)outp;
#pragma unroll
    for (int nt = 0; nt < 8; ++nt) {
      int o = Nstrip + nt * 16 + lo;
#pragma unroll
      for (int r = 0; r < 8; ++r) {
        int t = Mbase + 8 * hi + r;
        out[(size_t)t * Dc + o] = acc[nt].f[r];
      }
    }
  }
}

// ---------------- RoPE (interleaved pairs), in place on bf16 (bh,s,d) ----------------
__global__ void k_rope(u16t* __restrict__ Q, u16t* __restrict__ K) {
  u16t* X = blockIdx.y ? K : Q;
  int i = blockIdx.x * blockDim.x + threadIdx.x;
  int j = i & 63;
  int t = i >> 6;
  int s = t & (Sc - 1);
  float inv_freq = __expf(-(float)(2 * j) * (9.210340371976184f / 128.0f));
  float ang = (float)s * inv_freq;
  float c, sn;
  __sincosf(ang, &c, &sn);
  u16t* p = X + (size_t)t * DHc + 2 * j;
  float xe = bf2f(p[0]), xo = bf2f(p[1]);
  p[0] = f2bf(xe * c - xo * sn);
  p[1] = f2bf(xe * sn + xo * c);
}

// ---------------- Flash attention (causal), one wave = 16 q-rows ----------------
// Q,K: (bh,s,dh) bf16.  Vt: (bh,dh,s) bf16.  Out: (b,s,h,dh) bf16.
__global__ __launch_bounds__(256, 1) void k_attn(const u16t* __restrict__ Q,
                                                 const u16t* __restrict__ K,
                                                 const u16t* __restrict__ Vt,
                                                 u16t* __restrict__ Out) {
  const int lane = threadIdx.x & 31;
  const int wave = threadIdx.x >> 5;
  const int lo = lane & 15;
  const int hi = lane >> 4;
  const int bh = blockIdx.y;
  const int b = bh >> 4, h = bh & 15;
  const int qbase = blockIdx.x * 128 + wave * 16;

  const u16t* Qb = Q + (size_t)bh * Sc * DHc;
  const u16t* Kb = K + (size_t)bh * Sc * DHc;
  const u16t* Vb = Vt + (size_t)bh * DHc * Sc;

  // Q^T as B operand: lane = q column, k = 8*hi+0..7 and 16+8*hi+0..7 per lane-half
  Frag qf[4];
  {
    const u16t* qrow = Qb + (size_t)(qbase + lo) * DHc;
#pragma unroll
    for (int kk = 0; kk < 4; ++kk) {
      qf[kk].q[0] = *(const uint4*)(qrow + kk * 32 + 8 * hi);
      qf[kk].q[1] = *(const uint4*)(qrow + kk * 32 + 16 + 8 * hi);
    }
  }

  Acc acc[8];                    // O^T: 8 d-tiles of (16 d x 16 q)
#pragma unroll
  for (int i = 0; i < 8; ++i)
#pragma unroll
    for (int r = 0; r < 8; ++r) acc[i].f[r] = 0.0f;

  float m_i = -3.0e38f, l_i = 0.0f;
  const int qg = qbase + lo;

  for (int kvb = 0; kvb < qbase + 16; kvb += 32) {
    const u16t* kr0 = Kb + (size_t)(kvb + lo) * DHc;
    const u16t* kr1 = kr0 + 16 * DHc;

    // stage ALL K fragments; fence pins the 16 loads ahead of the WMMAs
    Frag kf0[4], kf1[4];
#pragma unroll
    for (int kk = 0; kk < 4; ++kk) {
      kf0[kk].q[0] = *(const uint4*)(kr0 + kk * 32 + 8 * hi);
      kf0[kk].q[1] = *(const uint4*)(kr0 + kk * 32 + 16 + 8 * hi);
      kf1[kk].q[0] = *(const uint4*)(kr1 + kk * 32 + 8 * hi);
      kf1[kk].q[1] = *(const uint4*)(kr1 + kk * 32 + 16 + 8 * hi);
    }
    sched_fence();

    Acc s0, s1;
#pragma unroll
    for (int r = 0; r < 8; ++r) { s0.f[r] = 0.0f; s1.f[r] = 0.0f; }
#pragma unroll
    for (int kk = 0; kk < 4; ++kk) {
      s0.v = wmma_bf16(kf0[kk].v, qf[kk].v, s0.v);
      s1.v = wmma_bf16(kf1[kk].v, qf[kk].v, s1.v);
    }

    // issue V-fragment loads now; fence keeps them ahead of the softmax VALU block
    Frag vf[8];
#pragma unroll
    for (int dt = 0; dt < 8; ++dt) {
      const u16t* vrow = Vb + (size_t)(dt * 16 + lo) * Sc + kvb;
      vf[dt].q[0] = *(const uint4*)(vrow + 16 * hi);
      vf[dt].q[1] = *(const uint4*)(vrow + 16 * hi + 8);
    }
    sched_fence();

    // scale + causal mask; online softmax (state per lane = per q row)
    float mx = -3.0e38f;
#pragma unroll
    for (int r = 0; r < 8; ++r) {
      int kv0 = kvb + 8 * hi + r;
      int kv1 = kv0 + 16;
      s0.f[r] = (kv0 <= qg) ? s0.f[r] * SCALE : -1.0e30f;
      s1.f[r] = (kv1 <= qg) ? s1.f[r] * SCALE : -1.0e30f;
      mx = fmaxf(mx, fmaxf(s0.f[r], s1.f[r]));
    }
    mx = fmaxf(mx, __shfl_xor(mx, 16, 32));
    float m_new = fmaxf(m_i, mx);
    float alpha = __expf(m_i - m_new);

    float p0[8], p1[8], rsum = 0.0f;
#pragma unroll
    for (int r = 0; r < 8; ++r) {
      p0[r] = __expf(s0.f[r] - m_new);
      p1[r] = __expf(s1.f[r] - m_new);
      rsum += p0[r] + p1[r];
    }
    rsum += __shfl_xor(rsum, 16, 32);
    l_i = l_i * alpha + rsum;
    m_i = m_new;

#pragma unroll
    for (int dt = 0; dt < 8; ++dt)
#pragma unroll
      for (int r = 0; r < 8; ++r) acc[dt].f[r] *= alpha;

    // P^T as B operand (32 kv x 16 q): lane needs kv = kvb + 16*hi + 0..15
    Frag pf;
#pragma unroll
    for (int r = 0; r < 8; ++r) {
      float x0 = __shfl_xor(p0[r], 16, 32);
      float x1 = __shfl_xor(p1[r], 16, 32);
      pf.u[r]     = f2bf(hi ? x1 : p0[r]);
      pf.u[r + 8] = f2bf(hi ? p1[r] : x0);
    }

#pragma unroll
    for (int dt = 0; dt < 8; ++dt)
      acc[dt].v = wmma_bf16(vf[dt].v, pf.v, acc[dt].v);
  }

  float inv = 1.0f / l_i;
  size_t obase = ((size_t)(b * Sc + qg) * Hc + h) * DHc;
#pragma unroll
  for (int dt = 0; dt < 8; ++dt) {
    uint4 w;
    w.x = pack2(acc[dt].f[0] * inv, acc[dt].f[1] * inv);
    w.y = pack2(acc[dt].f[2] * inv, acc[dt].f[3] * inv);
    w.z = pack2(acc[dt].f[4] * inv, acc[dt].f[5] * inv);
    w.w = pack2(acc[dt].f[6] * inv, acc[dt].f[7] * inv);
    *(uint4*)(Out + obase + dt * 16 + 8 * hi) = w;
  }
}

extern "C" void kernel_launch(void* const* d_in, const int* in_sizes, int n_in,
                              void* d_out, int out_size, void* d_ws, size_t ws_size,
                              hipStream_t stream) {
  (void)in_sizes; (void)n_in; (void)out_size; (void)ws_size;
  const float* x  = (const float*)d_in[0];
  const float* Wq = (const float*)d_in[1];
  const float* Wk = (const float*)d_in[2];
  const float* Wv = (const float*)d_in[3];
  const float* Wo = (const float*)d_in[4];

  const size_t NX = (size_t)Bc * Sc * Dc;   // 16,777,216
  const size_t NW = (size_t)Dc * Dc;        //  4,194,304

  char* ws = (char*)d_ws;
  u16t* xb   = (u16t*)ws;
  u16t* wqb  = (u16t*)(ws + NX * 2);
  u16t* wkb  = wqb + NW;
  u16t* wvb  = wkb + NW;
  u16t* wob  = wvb + NW;
  u16t* qb   = wob + NW;                     // (bh,s,d)
  u16t* kb   = qb + NX;                      // (bh,s,d)
  u16t* vt   = kb + NX;                      // (bh,d,s)
  u16t* attn = xb;                           // aliases xb (x dead by then)

  dim3 blk(256);
  k_cvt_bf16<<<(unsigned)(NX / 256), blk, 0, stream>>>(x,  xb,  (int)NX);
  k_cvt_bf16<<<(unsigned)(NW / 256), blk, 0, stream>>>(Wq, wqb, (int)NW);
  k_cvt_bf16<<<(unsigned)(NW / 256), blk, 0, stream>>>(Wk, wkb, (int)NW);
  k_cvt_bf16<<<(unsigned)(NW / 256), blk, 0, stream>>>(Wv, wvb, (int)NW);
  k_cvt_bf16<<<(unsigned)(NW / 256), blk, 0, stream>>>(Wo, wob, (int)NW);

  dim3 ggrid(Dc / 128, (Bc * Sc) / 128);     // (16, 64)
  k_gemm_nt<0><<<ggrid, blk, 0, stream>>>(xb, wqb, qb);
  k_gemm_nt<0><<<ggrid, blk, 0, stream>>>(xb, wkb, kb);
  k_gemm_nt<1><<<ggrid, blk, 0, stream>>>(xb, wvb, vt);

  dim3 rgrid((unsigned)((size_t)Bc * Hc * Sc * (DHc / 2) / 256), 2);
  k_rope<<<rgrid, blk, 0, stream>>>(qb, kb);

  dim3 agrid(Sc / 128, Bc * Hc);             // (16, 64)
  k_attn<<<agrid, blk, 0, stream>>>(qb, kb, vt, attn);

  k_gemm_nt<2><<<ggrid, blk, 0, stream>>>(attn, wob, d_out);
}
